// Decoder_14568529068261
// MI455X (gfx1250) — compile-verified, matrix operand order checked by reading, not run on hardware
//
#include <hip/hip_runtime.h>
#include <hip/hip_bf16.h>

// Problem constants baked from the reference: V,E,D,L,B,EX = 50000,128,256,400,64,200
#define Vv  50000
#define Ee  128
#define Dd  256
#define Ll  400
#define Bb  64
#define EXx 200

typedef __attribute__((ext_vector_type(16))) __bf16 v16bf;
typedef __attribute__((ext_vector_type(8)))  __bf16 v8bf;
typedef __attribute__((ext_vector_type(8)))  float  v8f;

// Native RNE convert: clang emits v_cvt_pk_bf16_f32.
__device__ __forceinline__ __bf16 f2bf(float f) { return static_cast<__bf16>(f); }

// Hardware reciprocal (v_rcp_f32, ~1 ulp) - avoids IEEE div expansion.
__device__ __forceinline__ float fast_rcp(float x) { return __builtin_amdgcn_rcpf(x); }

__device__ __forceinline__ float sigmoidf(float x) {
    return fast_rcp(1.f + __expf(-x));
}

// Branch-free tanh: exact saturation at +/-1, no exec-mask divergence, no div.
__device__ __forceinline__ float fast_tanh(float x) {
    float e = __expf(2.f * x);                 // +inf for large x -> result 1
    return 1.f - 2.f * fast_rcp(e + 1.f);      // e->0 for very negative x -> -1
}

// ---------------- Stage 1: x = [emb, prev_ctx] @ lin_w.T + lin_b  (64x256) ----
__global__ void k_x(const int* __restrict__ inp, const float* __restrict__ emb_w,
                    const float* __restrict__ prev_ctx, const float* __restrict__ lin_w,
                    const float* __restrict__ lin_b, float* __restrict__ x) {
    int b = blockIdx.x, j = threadIdx.x;
    const float* wr = lin_w + j * (Ee + 2 * Dd);
    const float* e  = emb_w + (long)inp[b] * Ee;
    const float* pc = prev_ctx + b * (2 * Dd);
    float acc = lin_b[j];
    for (int k = 0; k < Ee; ++k)      acc += e[k]  * wr[k];
    for (int k = 0; k < 2 * Dd; ++k)  acc += pc[k] * wr[Ee + k];
    x[b * Dd + j] = acc;
}

// ---------------- Stage 2: bridge reduce (branch on output_step in-kernel) ----
__global__ void k_reduce_hc(const int* __restrict__ step,
                            const float* __restrict__ hh, const float* __restrict__ hc,
                            const float* __restrict__ rhw, const float* __restrict__ rhb,
                            const float* __restrict__ rcw, const float* __restrict__ rcb,
                            float* __restrict__ h, float* __restrict__ c) {
    int b = blockIdx.x, j = threadIdx.x;
    if (step[0] == 0) {
        const float* h0 = hh + b * Dd;                 // layer 0
        const float* h1 = hh + Bb * Dd + b * Dd;       // layer 1
        const float* c0 = hc + b * Dd;
        const float* c1 = hc + Bb * Dd + b * Dd;
        const float* wh = rhw + (long)j * (2 * Dd);
        const float* wc = rcw + (long)j * (2 * Dd);
        float ah = rhb[j], ac = rcb[j];
        for (int k = 0; k < Dd; ++k) {
            ah += h0[k] * wh[k] + h1[k] * wh[Dd + k];
            ac += c0[k] * wc[k] + c1[k] * wc[Dd + k];
        }
        h[b * Dd + j] = fmaxf(ah, 0.f);
        c[b * Dd + j] = fmaxf(ac, 0.f);
    } else {
        h[b * Dd + j] = hh[b * Dd + j];
        c[b * Dd + j] = hc[b * Dd + j];
    }
}

// ---------------- Stage 3: LSTM cell + Q = tanh([h',c'] @ wq.T)/sqrt(D) -------
__global__ void k_lstm_q(const float* __restrict__ x, const float* __restrict__ h,
                         const float* __restrict__ c,
                         const float* __restrict__ w_ih, const float* __restrict__ w_hh,
                         const float* __restrict__ b_ih, const float* __restrict__ b_hh,
                         const float* __restrict__ wq_w, const float* __restrict__ wq_b,
                         float* __restrict__ h_new, float* __restrict__ c_new,
                         float* __restrict__ Qs) {
    __shared__ float sx[Dd], sh[Dd], shn[Dd], scn[Dd];
    int b = blockIdx.x, j = threadIdx.x;
    sx[j] = x[b * Dd + j];
    sh[j] = h[b * Dd + j];
    __syncthreads();
    float g4[4];
    #pragma unroll
    for (int g = 0; g < 4; ++g) {
        int r = g * Dd + j;
        const float* wi = w_ih + (long)r * Dd;
        const float* wh = w_hh + (long)r * Dd;
        float a = b_ih[r] + b_hh[r];
        for (int k = 0; k < Dd; ++k) a += sx[k] * wi[k] + sh[k] * wh[k];
        g4[g] = a;
    }
    float cv = c[b * Dd + j];
    float cn = sigmoidf(g4[1]) * cv + sigmoidf(g4[0]) * fast_tanh(g4[2]);
    float hn = sigmoidf(g4[3]) * fast_tanh(cn);
    h_new[b * Dd + j] = hn;
    c_new[b * Dd + j] = cn;
    shn[j] = hn; scn[j] = cn;
    __syncthreads();
    const float* wq = wq_w + (long)j * (2 * Dd);
    float a = wq_b[j];
    for (int k = 0; k < Dd; ++k) a += shn[k] * wq[k] + scn[k] * wq[Dd + k];
    Qs[b * Dd + j] = fast_tanh(a) * (1.0f / 16.0f);   // sqrt(256) = 16
}

// ---------------- Stage 4: fused K-projection + score, WMMA bf16 --------------
// One wave per 16-row (b,l) tile; 400 = 25*16 so tiles never cross batches.
// score[b,l] = sum_n tanh( wf[b,l,:] @ wk_w[n,:] + wk_b[n] ) * Q[b,n]/sqrt(D)
__global__ void __launch_bounds__(64) k_attn_score(
        const float* __restrict__ enc, const float* __restrict__ cov,
        const float* __restrict__ cov_w, const float* __restrict__ cov_b,
        const float* __restrict__ wk_w, const float* __restrict__ wk_b,
        const float* __restrict__ Qs, const float* __restrict__ pmask,
        float* __restrict__ score) {
    __shared__ __align__(16) __bf16 smA[2][16][2 * Dd];   // 32 KB: word_feature tiles (bf16)
    const int wid = threadIdx.x >> 5, lane = threadIdx.x & 31;
    const int tile = blockIdx.x * 2 + wid;                // 0..1599
    const int b = tile / 25, l0 = (tile % 25) * 16;

    // Phase 1: stage word_feature tile (16 x 512) into LDS as bf16
    const float* encB = enc + ((long)(b * Ll + l0)) * (2 * Dd);
    const float* covB = cov + b * Ll + l0;
    for (int t = lane; t < 16 * 2 * Dd; t += 32) {
        int row = t >> 9, col = t & (2 * Dd - 1);
        float wf = encB[(long)row * (2 * Dd) + col] + covB[row] * cov_w[col] + cov_b[col];
        smA[wid][row][col] = f2bf(wf);
    }
    __syncthreads();

    const int m_a  = lane & 15;
    const int koff = (lane & 16) >> 1;                    // 0 or 8 (ISA 16-bit A layout)
    float partial[8];
    #pragma unroll
    for (int m = 0; m < 8; ++m) partial[m] = 0.f;

    for (int nt = 0; nt < 16; ++nt) {                     // 256 output cols / 16
        const int n = nt * 16 + (lane & 15);
        const float q = Qs[b * Dd + n];
        const float biasn = wk_b[n];
        const float* wrow = wk_w + (long)n * (2 * Dd);
        v8f acc = {0.f, 0.f, 0.f, 0.f, 0.f, 0.f, 0.f, 0.f};
        for (int kc = 0; kc < 16; ++kc) {                 // K = 512 in chunks of 32
            const int k0 = kc * 32 + koff;
            v8bf alo = *(const v8bf*)&smA[wid][m_a][k0];
            v8bf ahi = *(const v8bf*)&smA[wid][m_a][k0 + 16];
            v16bf av, bv;
            #pragma unroll
            for (int i = 0; i < 8; ++i) {
                av[i]     = alo[i];
                av[8 + i] = ahi[i];
                bv[i]     = f2bf(wrow[k0 + i]);
                bv[8 + i] = f2bf(wrow[k0 + 16 + i]);
            }
            acc = __builtin_amdgcn_wmma_f32_16x16x32_bf16(
                      false, av, false, bv, (short)0, acc, false, false);
        }
        #pragma unroll
        for (int m = 0; m < 8; ++m) partial[m] += fast_tanh(acc[m] + biasn) * q;
    }
    // Reduce over the 16 lanes (distinct n) within each lane-half.
    #pragma unroll
    for (int off = 1; off < 16; off <<= 1) {
        #pragma unroll
        for (int m = 0; m < 8; ++m) partial[m] += __shfl_xor(partial[m], off, 32);
    }
    if ((lane & 15) == 0) {
        const int hi = (lane >> 4) * 8;                   // lanes>=16 hold rows M=8..15
        #pragma unroll
        for (int m = 0; m < 8; ++m) {
            int l = l0 + m + hi;
            float s = partial[m];
            s = (pmask[b * Ll + l] > 0.f) ? s : -1e18f;
            score[b * Ll + l] = s;
        }
    }
}

// ---------------- Stage 5: softmax over L + weighted context ------------------
__global__ void k_softmax_wc(const float* __restrict__ score, const float* __restrict__ enc,
                             float* __restrict__ att_ws, float* __restrict__ wc_ws,
                             float* __restrict__ att_out, float* __restrict__ wc_out) {
    __shared__ float sm[Ll];
    __shared__ float red[256];
    int b = blockIdx.x, t = threadIdx.x;
    float mx = -INFINITY;
    for (int l = t; l < Ll; l += 256) { float v = score[b * Ll + l]; sm[l] = v; mx = fmaxf(mx, v); }
    red[t] = mx; __syncthreads();
    for (int s = 128; s > 0; s >>= 1) { if (t < s) red[t] = fmaxf(red[t], red[t + s]); __syncthreads(); }
    float m = red[0]; __syncthreads();
    float sum = 0.f;
    for (int l = t; l < Ll; l += 256) { float e = __expf(sm[l] - m); sm[l] = e; sum += e; }
    red[t] = sum; __syncthreads();
    for (int s = 128; s > 0; s >>= 1) { if (t < s) red[t] += red[t + s]; __syncthreads(); }
    float inv = 1.f / red[0]; __syncthreads();
    for (int l = t; l < Ll; l += 256) {
        float a = sm[l] * inv; sm[l] = a;
        att_ws[b * Ll + l] = a; att_out[b * Ll + l] = a;
    }
    __syncthreads();
    for (int d = t; d < 2 * Dd; d += 256) {
        float acc = 0.f;
        const float* e = enc + (long)b * Ll * (2 * Dd) + d;
        for (int l = 0; l < Ll; ++l) acc += sm[l] * e[(long)l * (2 * Dd)];
        wc_ws[b * (2 * Dd) + d] = acc; wc_out[b * (2 * Dd) + d] = acc;
    }
}

// ---------------- Stage 6: p_gen ---------------------------------------------
__global__ void k_pgen(const float* __restrict__ x, const float* __restrict__ wc,
                       const float* __restrict__ h_new, const float* __restrict__ c_new,
                       const float* __restrict__ pgen_w, const float* __restrict__ pgen_b,
                       float* __restrict__ pgen_ws, float* __restrict__ pgen_out) {
    __shared__ float red[128];
    int b = blockIdx.x, t = threadIdx.x;
    float a = 0.f;
    for (int k = t; k < 5 * Dd; k += 128) {
        float v;
        if (k < Dd)            v = x[b * Dd + k];
        else if (k < 3 * Dd)   v = wc[b * (2 * Dd) + (k - Dd)];
        else if (k < 4 * Dd)   v = h_new[b * Dd + (k - 3 * Dd)];
        else                   v = c_new[b * Dd + (k - 4 * Dd)];
        a += v * pgen_w[k];
    }
    red[t] = a; __syncthreads();
    for (int s = 64; s > 0; s >>= 1) { if (t < s) red[t] += red[t + s]; __syncthreads(); }
    if (t == 0) {
        float p = sigmoidf(red[0] + pgen_b[0]);
        pgen_ws[b] = p; pgen_out[b] = p;
    }
}

// ---------------- Stage 7: pregen = [wc, h'] @ pre_w.T + pre_b ----------------
__global__ void k_pregen(const float* __restrict__ wc, const float* __restrict__ h_new,
                         const float* __restrict__ pre_w, const float* __restrict__ pre_b,
                         float* __restrict__ pregen) {
    int b = blockIdx.x, j = threadIdx.x;
    const float* wr = pre_w + (long)j * (3 * Dd);
    float a = pre_b[j];
    for (int k = 0; k < 2 * Dd; ++k) a += wc[b * (2 * Dd) + k] * wr[k];
    for (int k = 0; k < Dd; ++k)     a += h_new[b * Dd + k] * wr[2 * Dd + k];
    pregen[b * Dd + j] = a;
}

// ---------------- Stage 8: vocab logits GEMM, WMMA bf16 -----------------------
// logits(64,50000) = pregen(64,256) @ gen_w(50000,256).T + gen_b
__global__ void __launch_bounds__(128) k_vocab(const float* __restrict__ pregen,
                                               const float* __restrict__ gen_w,
                                               const float* __restrict__ gen_b,
                                               float* __restrict__ logits) {
    const int wid = threadIdx.x >> 5, lane = threadIdx.x & 31;
    const int task = blockIdx.x * 4 + wid;           // 12500 wave tiles
    const int mtile = task & 3, ntile = task >> 2;
    const int m_a = lane & 15, koff = (lane & 16) >> 1;
    const int n = ntile * 16 + (lane & 15);
    const float* arow = pregen + (long)(mtile * 16 + m_a) * Dd;
    const float* brow = gen_w + (long)n * Dd;
    v8f acc = {0.f, 0.f, 0.f, 0.f, 0.f, 0.f, 0.f, 0.f};
    for (int kc = 0; kc < 8; ++kc) {                 // K = 256 in chunks of 32
        const int k0 = kc * 32 + koff;
        v16bf av, bv;
        #pragma unroll
        for (int i = 0; i < 8; ++i) {
            av[i]     = f2bf(arow[k0 + i]);
            av[8 + i] = f2bf(arow[k0 + 16 + i]);
            bv[i]     = f2bf(brow[k0 + i]);
            bv[8 + i] = f2bf(brow[k0 + 16 + i]);
        }
        acc = __builtin_amdgcn_wmma_f32_16x16x32_bf16(
                  false, av, false, bv, (short)0, acc, false, false);
    }
    const float bias = gen_b[n];
    const int hi = (lane >> 4) * 8;
    #pragma unroll
    for (int r = 0; r < 8; ++r) {
        int m = mtile * 16 + r + hi;
        logits[(long)m * Vv + n] = acc[r] + bias;
    }
}

// ---------------- Stage 9: per-row max & sum(exp) over 50000 ------------------
__global__ void k_rowstats(const float* __restrict__ logits,
                           float* __restrict__ rowmax, float* __restrict__ rowsum) {
    __shared__ float red[256];
    int b = blockIdx.x, t = threadIdx.x;
    const float* row = logits + (long)b * Vv;
    float mx = -INFINITY;
    for (int i = t; i < Vv; i += 256) mx = fmaxf(mx, row[i]);
    red[t] = mx; __syncthreads();
    for (int s = 128; s > 0; s >>= 1) { if (t < s) red[t] = fmaxf(red[t], red[t + s]); __syncthreads(); }
    float m = red[0]; __syncthreads();
    float sum = 0.f;
    for (int i = t; i < Vv; i += 256) sum += __expf(row[i] - m);
    red[t] = sum; __syncthreads();
    for (int s = 128; s > 0; s >>= 1) { if (t < s) red[t] += red[t + s]; __syncthreads(); }
    if (t == 0) { rowmax[b] = m; rowsum[b] = red[0]; }
}

// ---------------- Stage 10: base final_dist (vocab part + EX zeros) -----------
__global__ void k_final_base(const float* __restrict__ logits, const float* __restrict__ rowmax,
                             const float* __restrict__ rowsum, const float* __restrict__ pgen,
                             float* __restrict__ fin) {
    int idx = blockIdx.x * 256 + threadIdx.x;
    if (idx >= Bb * (Vv + EXx)) return;
    int b = idx / (Vv + EXx), v = idx % (Vv + EXx);
    float o = 0.f;
    if (v < Vv) {
        // exp / rowsum via precise scale: rowsum is O(V), rcp fine for softmax
        o = pgen[b] * __expf(logits[(long)b * Vv + v] - rowmax[b]) / rowsum[b];
    }
    fin[idx] = o;
}

// ---------------- Stage 11: copy-mechanism scatter-add ------------------------
__global__ void k_scatter(const int* __restrict__ ext, const float* __restrict__ att,
                          const float* __restrict__ pgen, float* __restrict__ fin) {
    int idx = blockIdx.x * 256 + threadIdx.x;
    if (idx >= Bb * Ll) return;
    int b = idx / Ll;
    atomicAdd(&fin[(long)b * (Vv + EXx) + ext[idx]], (1.f - pgen[b]) * att[idx]);
}

// ---------------- Stage 12: pack remaining outputs ----------------------------
__global__ void k_pack(const float* __restrict__ h_new, const float* __restrict__ c_new,
                       const float* __restrict__ coverage, float* __restrict__ out,
                       long offH, long offC, long offCov) {
    int idx = blockIdx.x * 256 + threadIdx.x;
    const int NH = Bb * Dd;          // 16384
    const int NC = Bb * Ll;          // 25600
    if (idx < NH)                 out[offH + idx] = h_new[idx];
    else if (idx < 2 * NH)        out[offC + (idx - NH)] = c_new[idx - NH];
    else if (idx < 2 * NH + NC)   out[offCov + (idx - 2 * NH)] = coverage[idx - 2 * NH];
}

extern "C" void kernel_launch(void* const* d_in, const int* in_sizes, int n_in,
                              void* d_out, int out_size, void* d_ws, size_t ws_size,
                              hipStream_t stream) {
    (void)in_sizes; (void)n_in; (void)out_size; (void)ws_size;
    // Inputs in setup_inputs() dict order
    const int*   inp       = (const int*)  d_in[0];
    const float* hidden_h  = (const float*)d_in[1];
    const float* hidden_c  = (const float*)d_in[2];
    const float* enc       = (const float*)d_in[3];
    const float* prev_ctx  = (const float*)d_in[4];
    const int*   ext       = (const int*)  d_in[5];
    const float* coverage  = (const float*)d_in[6];
    const float* pmask     = (const float*)d_in[7];
    /* d_in[8] = ex_size (200) */
    const int*   out_step  = (const int*)  d_in[9];
    const float* emb_w     = (const float*)d_in[10];
    const float* lin_w     = (const float*)d_in[11];
    const float* lin_b     = (const float*)d_in[12];
    const float* w_ih      = (const float*)d_in[13];
    const float* w_hh      = (const float*)d_in[14];
    const float* b_ih      = (const float*)d_in[15];
    const float* b_hh      = (const float*)d_in[16];
    const float* red_h_w   = (const float*)d_in[17];
    const float* red_h_b   = (const float*)d_in[18];
    const float* red_c_w   = (const float*)d_in[19];
    const float* red_c_b   = (const float*)d_in[20];
    const float* wq_w      = (const float*)d_in[21];
    const float* wq_b      = (const float*)d_in[22];
    const float* cov_w     = (const float*)d_in[23];
    const float* cov_b     = (const float*)d_in[24];
    const float* wk_w      = (const float*)d_in[25];
    const float* wk_b      = (const float*)d_in[26];
    const float* pre_w     = (const float*)d_in[27];
    const float* pre_b     = (const float*)d_in[28];
    const float* gen_w     = (const float*)d_in[29];
    const float* gen_b     = (const float*)d_in[30];
    const float* pgen_w    = (const float*)d_in[31];
    const float* pgen_b    = (const float*)d_in[32];

    float* out = (float*)d_out;
    // Output layout (flat, return order)
    const long OFF_H   = (long)Bb * (Vv + EXx);          // 3,212,800
    const long OFF_C   = OFF_H  + Bb * Dd;
    const long OFF_WC  = OFF_C  + Bb * Dd;
    const long OFF_ATT = OFF_WC + Bb * 2 * Dd;
    const long OFF_COV = OFF_ATT + Bb * Ll;
    const long OFF_PG  = OFF_COV + Bb * Ll;

    // Workspace layout (floats); total ~13.6 MB
    float* ws = (float*)d_ws;
    float* ws_x      = ws;                         // 16384
    float* ws_h      = ws_x + Bb * Dd;             // 16384
    float* ws_c      = ws_h + Bb * Dd;             // 16384
    float* ws_hn     = ws_c + Bb * Dd;             // 16384
    float* ws_cn     = ws_hn + Bb * Dd;            // 16384
    float* ws_q      = ws_cn + Bb * Dd;            // 16384
    float* ws_score  = ws_q + Bb * Dd;             // 25600
    float* ws_att    = ws_score + Bb * Ll;         // 25600
    float* ws_wc     = ws_att + Bb * Ll;           // 32768
    float* ws_pregen = ws_wc + Bb * 2 * Dd;        // 16384
    float* ws_pgen   = ws_pregen + Bb * Dd;        // 64
    float* ws_rmax   = ws_pgen + Bb;               // 64
    float* ws_rsum   = ws_rmax + Bb;               // 64
    float* ws_logits = ws_rsum + Bb;               // 3,200,000

    k_x<<<Bb, Dd, 0, stream>>>(inp, emb_w, prev_ctx, lin_w, lin_b, ws_x);
    k_reduce_hc<<<Bb, Dd, 0, stream>>>(out_step, hidden_h, hidden_c,
                                       red_h_w, red_h_b, red_c_w, red_c_b, ws_h, ws_c);
    k_lstm_q<<<Bb, Dd, 0, stream>>>(ws_x, ws_h, ws_c, w_ih, w_hh, b_ih, b_hh,
                                    wq_w, wq_b, ws_hn, ws_cn, ws_q);
    // 1600 (b,l) tiles of 16 rows, 2 waves/block
    k_attn_score<<<800, 64, 0, stream>>>(enc, coverage, cov_w, cov_b, wk_w, wk_b,
                                         ws_q, pmask, ws_score);
    k_softmax_wc<<<Bb, 256, 0, stream>>>(ws_score, enc, ws_att, ws_wc,
                                         out + OFF_ATT, out + OFF_WC);
    k_pgen<<<Bb, 128, 0, stream>>>(ws_x, ws_wc, ws_hn, ws_cn, pgen_w, pgen_b,
                                   ws_pgen, out + OFF_PG);
    k_pregen<<<Bb, Dd, 0, stream>>>(ws_wc, ws_hn, pre_w, pre_b, ws_pregen);
    // 4 m-tiles x 3125 n-tiles, 4 waves/block
    k_vocab<<<3125, 128, 0, stream>>>(ws_pregen, gen_w, gen_b, ws_logits);
    k_rowstats<<<Bb, 256, 0, stream>>>(ws_logits, ws_rmax, ws_rsum);
    {
        int n = Bb * (Vv + EXx);
        k_final_base<<<(n + 255) / 256, 256, 0, stream>>>(ws_logits, ws_rmax, ws_rsum,
                                                          ws_pgen, out);
    }
    k_scatter<<<(Bb * Ll + 255) / 256, 256, 0, stream>>>(ext, ws_att, ws_pgen, out);
    {
        int n = 2 * Bb * Dd + Bb * Ll;
        k_pack<<<(n + 255) / 256, 256, 0, stream>>>(ws_hn, ws_cn, coverage, out,
                                                    OFF_H, OFF_C, OFF_COV);
    }
}
